// Custom_Model_Zentorch_Qlinear_X3_25649544691851
// MI455X (gfx1250) — 8-bit matrix operand order repaired and checked by reading, compiled, not run on hardware
//
#include <hip/hip_runtime.h>
#include <hip/hip_bf16.h>


typedef int v8i __attribute__((ext_vector_type(8)));

union Frag {
    v8i  v;
    int4 q[2];
    int2 d[4];
};

static constexpr int M = 8192;
static constexpr int K = 4096;
static constexpr int N = 4096;
static constexpr int BK = 64;
static constexpr int NKT = K / BK;

__global__ __launch_bounds__(256) void pack_w_kernel(const int* __restrict__ w,
                                                     signed char* __restrict__ w8) {
    int i = blockIdx.x * 256 + threadIdx.x;
    if (i < (N * K) / 4) {
        int4 v = reinterpret_cast<const int4*>(w)[i];
        int packed = (v.x & 0xFF) | ((v.y & 0xFF) << 8) |
                     ((v.z & 0xFF) << 16) | ((v.w & 0xFF) << 24);
        reinterpret_cast<int*>(w8)[i] = packed;
    }
}

__global__ __launch_bounds__(256) void quant_in_kernel(const float* __restrict__ x,
                                                       const float* __restrict__ s_in_p,
                                                       const int*   __restrict__ zp_p,
                                                       signed char* __restrict__ qx) {
    const float inv_s = 1.0f / s_in_p[0];
    const float zpf   = (float)zp_p[0];
    const float lo = -zpf, hi = 255.0f - zpf;
    int i = blockIdx.x * 256 + threadIdx.x;
    if (i < (M * K) / 4) {
        float4 v = reinterpret_cast<const float4*>(x)[i];
        int q0 = (int)fminf(fmaxf(rintf(v.x * inv_s), lo), hi);
        int q1 = (int)fminf(fmaxf(rintf(v.y * inv_s), lo), hi);
        int q2 = (int)fminf(fmaxf(rintf(v.z * inv_s), lo), hi);
        int q3 = (int)fminf(fmaxf(rintf(v.w * inv_s), lo), hi);
        int packed = (q0 & 0xFF) | ((q1 & 0xFF) << 8) |
                     ((q2 & 0xFF) << 16) | ((q3 & 0xFF) << 24);
        reinterpret_cast<int*>(qx)[i] = packed;
    }
}

template <bool STORE_INT8>
__global__ __launch_bounds__(256)
void qgemm_i8_kernel(const signed char* __restrict__ qx,
                     const signed char* __restrict__ w8,
                     const float* __restrict__ s_w,
                     const float* __restrict__ bias,
                     const float* __restrict__ s_in_p,
                     const int*   __restrict__ zp_p,
                     signed char* __restrict__ qout,
                     float*       __restrict__ fout)
{
    __shared__ int4 As4[2][128][4];
    __shared__ int4 Bs4[2][128][4];

    const int t       = threadIdx.x;
    const int lane    = t & 31;
    const int wave    = t >> 5;
    const int wm      = (wave & 3) * 32;
    const int wn      = (wave >> 2) * 64;
    const int lh      = lane >> 4;
    const int ln      = lane & 15;
    const int m_block = blockIdx.y * 128;
    const int n_block = blockIdx.x * 128;

    const int ldRow = t >> 1;
    const int ldOff = (t & 1) * 32;
    const signed char* aGlob = qx + (size_t)(m_block + ldRow) * K + ldOff;
    const signed char* bGlob = w8 + (size_t)(n_block + ldRow) * K + ldOff;

    const signed char* Abase = reinterpret_cast<const signed char*>(As4);
    const signed char* Bbase = reinterpret_cast<const signed char*>(Bs4);

    v8i acc[2][4];
#pragma unroll
    for (int mt = 0; mt < 2; ++mt)
#pragma unroll
        for (int nt = 0; nt < 4; ++nt)
            acc[mt][nt] = (v8i){};

    {
        int4 ar0 = *reinterpret_cast<const int4*>(aGlob);
        int4 ar1 = *reinterpret_cast<const int4*>(aGlob + 16);
        int4 br0 = *reinterpret_cast<const int4*>(bGlob);
        int4 br1 = *reinterpret_cast<const int4*>(bGlob + 16);
        int4* ad = &As4[0][ldRow][ldOff >> 4];
        int4* bd = &Bs4[0][ldRow][ldOff >> 4];
        ad[0] = ar0; ad[1] = ar1;
        bd[0] = br0; bd[1] = br1;
    }

    for (int kt = 0; kt < NKT; ++kt) {
        const int cur = kt & 1;

        __syncthreads();

        int4 ar0, ar1, br0, br1;
        const bool have_next = (kt + 1 < NKT);
        if (have_next) {
            const signed char* ag = aGlob + (size_t)(kt + 1) * BK;
            const signed char* bg = bGlob + (size_t)(kt + 1) * BK;
            ar0 = *reinterpret_cast<const int4*>(ag);
            ar1 = *reinterpret_cast<const int4*>(ag + 16);
            br0 = *reinterpret_cast<const int4*>(bg);
            br1 = *reinterpret_cast<const int4*>(bg + 16);
            if (kt + 2 < NKT) {
                __builtin_prefetch(aGlob + (size_t)(kt + 2) * BK, 0, 1);
                __builtin_prefetch(bGlob + (size_t)(kt + 2) * BK, 0, 1);
            }
        }

        Frag af[2];
#pragma unroll
        for (int mt = 0; mt < 2; ++mt) {
            const int4* row = reinterpret_cast<const int4*>(
                Abase + cur * 8192 + (wm + mt * 16 + ln) * 64);
            af[mt].q[0] = row[0 + lh];
            af[mt].q[1] = row[2 + lh];
        }
        Frag bf[4];
#pragma unroll
        for (int nt = 0; nt < 4; ++nt) {
            const int4* row = reinterpret_cast<const int4*>(
                Bbase + cur * 8192 + (wn + nt * 16 + ln) * 64);
            bf[nt].q[0] = row[0 + lh];
            bf[nt].q[1] = row[2 + lh];
        }

#pragma unroll
        for (int mt = 0; mt < 2; ++mt)
#pragma unroll
            for (int nt = 0; nt < 4; ++nt)
                acc[mt][nt] = __builtin_amdgcn_wmma_i32_16x16x64_iu8(
                     true, af[mt].v,  true, bf[nt].v,
                    acc[mt][nt],  false,  false);

        if (have_next) {
            const int nxt = cur ^ 1;
            int4* ad = &As4[nxt][ldRow][ldOff >> 4];
            int4* bd = &Bs4[nxt][ldRow][ldOff >> 4];
            ad[0] = ar0; ad[1] = ar1;
            bd[0] = br0; bd[1] = br1;
        }
    }

    const float sin_v = s_in_p[0];
    const float inv_s = 1.0f / sin_v;
    const float zpf   = (float)zp_p[0];
    const float lo = -zpf, hi = 255.0f - zpf;

#pragma unroll
    for (int mt = 0; mt < 2; ++mt) {
#pragma unroll
        for (int nt = 0; nt < 4; ++nt) {
            const int n  = n_block + wn + nt * 16 + ln;
            const float scale = sin_v * s_w[n];
            const float bb    = bias[n];
#pragma unroll
            for (int j = 0; j < 8; ++j) {
                const int m = m_block + wm + mt * 16 + j + lh * 8;
                const float val = scale * (float)acc[mt][nt][j] + bb;
                const size_t idx = (size_t)m * N + n;
                if (STORE_INT8) {
                    float qf = fminf(fmaxf(rintf(val * inv_s), lo), hi);
                    qout[idx] = (signed char)(int)qf;
                } else {
                    fout[idx] = val;
                }
            }
        }
    }
}

extern "C" void kernel_launch(void* const* d_in, const int* in_sizes, int n_in,
                              void* d_out, int out_size, void* d_ws, size_t ws_size,
                              hipStream_t stream) {
    const float* x    = (const float*)d_in[0];
    const int*   w    = (const int*)d_in[1];
    const float* bias = (const float*)d_in[2];
    const float* s_in = (const float*)d_in[3];
    const int*   zp   = (const int*)d_in[4];
    const float* s_w  = (const float*)d_in[5];
    float* out = (float*)d_out;

    signed char* w8  = (signed char*)d_ws;
    signed char* qxA = w8  + (size_t)N * K;
    signed char* qxB = qxA + (size_t)M * K;

    pack_w_kernel<<<(N * K / 4 + 255) / 256, 256, 0, stream>>>(w, w8);
    quant_in_kernel<<<(M * K / 4 + 255) / 256, 256, 0, stream>>>(x, s_in, zp, qxA);

    dim3 grid(N / 128, M / 128);
    dim3 block(256);
    qgemm_i8_kernel<true ><<<grid, block, 0, stream>>>(qxA, w8, s_w, bias, s_in, zp, qxB, nullptr);
    qgemm_i8_kernel<true ><<<grid, block, 0, stream>>>(qxB, w8, s_w, bias, s_in, zp, qxA, nullptr);
    qgemm_i8_kernel<false><<<grid, block, 0, stream>>>(qxA, w8, s_w, bias, s_in, zp, nullptr, out);
}
